// GCNBlock_77927886618861
// MI455X (gfx1250) — matrix operand order repaired and checked by reading, not compiled, not run on hardware
//
#include <hip/hip_runtime.h>

typedef __attribute__((ext_vector_type(16))) __bf16 v16bf;
typedef __attribute__((ext_vector_type(8)))  float  v8f;

namespace {
constexpr int Bn = 8, Kn = 5, Cn = 64, Hn = 8192, DNn = 192;
constexpr int HLR = Hn / 2;                 // 4096
constexpr float RES_SCALE = 0.1f;
constexpr int WAVES = 8;                    // waves per block
constexpr int COLS_PER_BLOCK = WAVES * 16;  // 128 h-columns per block
constexpr int NT_W1 = 24;                   // w1: 4 mt x 6 kt tiles
constexpr int NT_WR = 8;                    // wr1/wr2: 4 mt x 2 kt tiles
constexpr int TILE_ELEMS = 512;             // 32 lanes x 16 bf16
constexpr int N_TILES = NT_W1 + 2 * NT_WR;  // 40
constexpr size_t LR_ELEMS  = (size_t)Bn * Cn * HLR; // 2,097,152
}

// Convert a 64x16 f32 result held as 4 C-layout accumulator tiles into the
// two 32x16 bf16 B-operand tiles for the next WMMA (K=64). Only a half-wave
// swap (shfl_xor 16) is needed.
__device__ __forceinline__ void cacc_to_bop(const v8f* acc, int half, v16bf* out) {
#pragma unroll
  for (int kt = 0; kt < 2; ++kt) {
#pragma unroll
    for (int r = 0; r < 8; ++r) {
      float lo  = acc[2 * kt][r];
      float hi  = acc[2 * kt + 1][r];
      float tlo = __shfl_xor(lo, 16, 32);
      float thi = __shfl_xor(hi, 16, 32);
      // half==0 lane: e[0..7]=rows kt*32+r (own lo), e[8..15]=rows kt*32+8+r (partner lo)
      // half==1 lane: e[0..7]=rows kt*32+16+r (partner hi), e[8..15]=rows kt*32+24+r (own hi)
      out[kt][r]     = (__bf16)(half ? thi : lo);
      out[kt][r + 8] = (__bf16)(half ? hi : tlo);
    }
  }
}

__global__ __launch_bounds__(256, 2) void gcn_fused_kernel(
    const float* __restrict__ knn_hr, const float* __restrict__ dp,
    const float* __restrict__ w1,  const float* __restrict__ b1,
    const float* __restrict__ wr1, const float* __restrict__ br1,
    const float* __restrict__ wr2, const float* __restrict__ br2,
    const float* __restrict__ w3,  const float* __restrict__ b3,
    float* __restrict__ out_knn)
{
  __shared__ __align__(32) __bf16 sW[N_TILES * TILE_ELEMS];  // 40 KB
  __shared__ float sB1[64], sBr1[64], sBr2[64], sW3[64], sB3[1];

  const int tid = threadIdx.x;

  // ---- pack weights into WMMA A-operand bf16 layout in LDS ----
  // This code's 16x32 A tile (contiguous k, as B uses): lane<16 holds row M=lane, elems {K0..15};
  // lane>=16 holds row M=lane-16, elems {K16..31}.
  for (int idx = tid; idx < N_TILES * TILE_ELEMS; idx += 256) {
    int tile = idx >> 9, e = idx & 511;
    int ln = e >> 4, j = e & 15;
    int hf = ln >> 4, mrow = ln & 15;
    int kk = (hf << 4) + j;   // K within 32-wide tile
    float v;
    if (tile < NT_W1) {
      int mt = tile / 6, kt = tile % 6;
      v = w1[(mt * 16 + mrow) * DNn + kt * 32 + kk];
    } else if (tile < NT_W1 + NT_WR) {
      int t = tile - NT_W1; int mt = t >> 1, kt = t & 1;
      v = wr1[(mt * 16 + mrow) * 64 + kt * 32 + kk];
    } else {
      int t = tile - NT_W1 - NT_WR; int mt = t >> 1, kt = t & 1;
      v = wr2[(mt * 16 + mrow) * 64 + kt * 32 + kk];
    }
    sW[idx] = (__bf16)v;
  }
  if (tid < 64) { sB1[tid] = b1[tid]; sBr1[tid] = br1[tid]; sBr2[tid] = br2[tid]; sW3[tid] = w3[tid]; }
  if (tid == 0) sB3[0] = b3[0];
  __syncthreads();

  const int wave = tid >> 5, lane = tid & 31;
  const int half = lane >> 4, n = lane & 15;
  const int blocksPerB = Hn / COLS_PER_BLOCK;                 // 64
  const int b  = blockIdx.x / blocksPerB;
  const int h0 = (blockIdx.x % blocksPerB) * COLS_PER_BLOCK + wave * 16;

  float lg[Kn];  // logits for this lane's column (h0+n), one per neighbor k

#pragma unroll 1
  for (int k = 0; k < Kn; ++k) {
    // Defeat LICM: without this, all 40 weight tiles (320 VGPRs) get hoisted
    // out of the k-loop, blowing past 256 VGPRs and spilling to scratch in
    // the hot loop. Re-reading each tile from LDS per use is far cheaper.
    asm volatile("" ::: "memory");

    // ---- GEMM1: X[64x16] = w1[64x192] * D[192x16]  (bf16 WMMA, f32 acc) ----
    v8f accX[4] = {};
    const float* dcol = dp + ((size_t)(b * Kn + k) * DNn) * Hn + h0 + n;
#pragma unroll
    for (int kt = 0; kt < 6; ++kt) {
      v16bf bm;
      const int kr0 = kt * 32 + half * 16;
#pragma unroll
      for (int j = 0; j < 16; ++j)
        bm[j] = (__bf16)dcol[(size_t)(kr0 + j) * Hn];   // 64B coalesced per half-wave
#pragma unroll
      for (int mt = 0; mt < 4; ++mt) {
        v16bf am = *(const v16bf*)(sW + (mt * 6 + kt) * TILE_ELEMS + lane * 16);
        accX[mt] = __builtin_amdgcn_wmma_f32_16x16x32_bf16(
            false, am, false, bm, (short)0, accX[mt], false, false);
      }
    }
#pragma unroll
    for (int mt = 0; mt < 4; ++mt)
#pragma unroll
      for (int r = 0; r < 8; ++r)
        accX[mt][r] += sB1[mt * 16 + half * 8 + r];

    v16bf xb[2];
    cacc_to_bop(accX, half, xb);

    // ---- GEMM2: R = relu(wr1 * X + br1) ----
    v8f accR[4] = {};
#pragma unroll
    for (int kt = 0; kt < 2; ++kt)
#pragma unroll
      for (int mt = 0; mt < 4; ++mt) {
        v16bf am = *(const v16bf*)(sW + (NT_W1 + mt * 2 + kt) * TILE_ELEMS + lane * 16);
        accR[mt] = __builtin_amdgcn_wmma_f32_16x16x32_bf16(
            false, am, false, xb[kt], (short)0, accR[mt], false, false);
      }
#pragma unroll
    for (int mt = 0; mt < 4; ++mt)
#pragma unroll
      for (int r = 0; r < 8; ++r)
        accR[mt][r] = fmaxf(accR[mt][r] + sBr1[mt * 16 + half * 8 + r], 0.f);

    v16bf rb[2];
    cacc_to_bop(accR, half, rb);

    // ---- GEMM3: R2 = wr2 * R ;  X += 0.1*(R2 + br2) ----
    v8f acc2[4] = {};
#pragma unroll
    for (int kt = 0; kt < 2; ++kt)
#pragma unroll
      for (int mt = 0; mt < 4; ++mt) {
        v16bf am = *(const v16bf*)(sW + (NT_W1 + NT_WR + mt * 2 + kt) * TILE_ELEMS + lane * 16);
        acc2[mt] = __builtin_amdgcn_wmma_f32_16x16x32_bf16(
            false, am, false, rb[kt], (short)0, acc2[mt], false, false);
      }
#pragma unroll
    for (int mt = 0; mt < 4; ++mt)
#pragma unroll
      for (int r = 0; r < 8; ++r)
        accX[mt][r] += RES_SCALE * (acc2[mt][r] + sBr2[mt * 16 + half * 8 + r]);

    // ---- logits = w3 . X + b3 (reduce across half-waves) ----
    float p = 0.f;
#pragma unroll
    for (int mt = 0; mt < 4; ++mt)
#pragma unroll
      for (int r = 0; r < 8; ++r)
        p += sW3[mt * 16 + half * 8 + r] * accX[mt][r];
    p += __shfl_xor(p, 16, 32);
    lg[k] = p + sB3[0];
  }

  // ---- softmax over k (registers, both halves hold duplicates) ----
  float mx = lg[0];
#pragma unroll
  for (int k = 1; k < Kn; ++k) mx = fmaxf(mx, lg[k]);
  float wk[Kn], s = 0.f;
#pragma unroll
  for (int k = 0; k < Kn; ++k) { wk[k] = __expf(lg[k] - mx); s += wk[k]; }
  float inv = 1.f / s;
#pragma unroll
  for (int k = 0; k < Kn; ++k) wk[k] *= inv;

  // ---- weighted sum of knn_hr over k; write knn ----
  const float* khb = knn_hr + (size_t)b * (Kn * Cn) * Hn + h0 + n;
  float* knnb = out_knn + (size_t)b * Cn * Hn + h0 + n;
#pragma unroll 1
  for (int c0 = 0; c0 < Cn; c0 += 2) {
    int c = c0 + half;          // two channels per pass, one per half-wave
    float acc = 0.f;
#pragma unroll
    for (int k = 0; k < Kn; ++k)
      acc += wk[k] * khb[(size_t)(k * Cn + c) * Hn];
    knnb[(size_t)c * Hn] = acc;
  }
}

// knn_lr[b,c,j] = 0.5*(knn[b,c,j] + knn[b,c,j+4096])  (reshape(b,c,2,H/2).mean(2))
__global__ __launch_bounds__(256) void downsample_kernel(
    const float* __restrict__ knn, float* __restrict__ lr)
{
  size_t idx = (size_t)blockIdx.x * blockDim.x + threadIdx.x;
  if (idx >= LR_ELEMS) return;
  size_t bc = idx >> 12;              // /4096
  size_t j  = idx & (HLR - 1);
  const float* p = knn + bc * Hn + j;
  lr[idx] = 0.5f * (p[0] + p[HLR]);
}

extern "C" void kernel_launch(void* const* d_in, const int* in_sizes, int n_in,
                              void* d_out, int out_size, void* d_ws, size_t ws_size,
                              hipStream_t stream) {
  const float* knn_hr = (const float*)d_in[0];
  const float* dp     = (const float*)d_in[1];
  const float* w1     = (const float*)d_in[2];
  const float* b1     = (const float*)d_in[3];
  const float* wr1    = (const float*)d_in[4];
  const float* br1    = (const float*)d_in[5];
  const float* wr2    = (const float*)d_in[6];
  const float* br2    = (const float*)d_in[7];
  const float* w3     = (const float*)d_in[8];
  const float* b3     = (const float*)d_in[9];

  float* out_lr  = (float*)d_out;            // [8,64,4096]
  float* out_knn = out_lr + LR_ELEMS;        // [8,64,8192]

  dim3 grid(Bn * (Hn / COLS_PER_BLOCK));     // 512 blocks
  dim3 block(WAVES * 32);                    // 256 threads = 8 waves
  gcn_fused_kernel<<<grid, block, 0, stream>>>(knn_hr, dp, w1, b1, wr1, br1,
                                               wr2, br2, w3, b3, out_knn);

  dim3 grid2((unsigned)((LR_ELEMS + 255) / 256));
  downsample_kernel<<<grid2, 256, 0, stream>>>(out_knn, out_lr);
}
